// CrossAttention_78451872629145
// MI455X (gfx1250) — matrix operand order repaired and checked by reading, not compiled, not run on hardware
//
#include <hip/hip_runtime.h>
#include <math.h>

#define BATCH 4
#define NQ 1024
#define NKV 2048
#define NHEAD 8
#define DHEAD 288            // 16 mv-ch * 16 comp + 32 scalars
#define TOT_MV 128
#define TOT_S 256

typedef __attribute__((ext_vector_type(16))) __bf16 v16bf;
typedef __attribute__((ext_vector_type(8)))  float  v8f;

#if defined(__AMDGCN__) && __has_builtin(__builtin_amdgcn_tensor_load_to_lds) && \
    __has_builtin(__builtin_amdgcn_s_wait_tensorcnt)
#define HAVE_TDM 1
#else
#define HAVE_TDM 0
#endif

#if HAVE_TDM
typedef unsigned int u32x4 __attribute__((ext_vector_type(4)));
typedef int          i32x4 __attribute__((ext_vector_type(4)));
typedef int          i32x8 __attribute__((ext_vector_type(8)));

// Issue one TDM descriptor: D# group0 per ISA 8.3 (count=1, lds_addr in
// bits[63:32], global_addr in bits[120:64], type=2 in bits[127:126]);
// group1 passed in pre-built (data_size / dims / strides per ISA 8.4).
// This toolchain exposes the 6-arg builtin (groups 2/3 + trailing group,
// all zero for <=2D tensors).
__device__ __forceinline__ void tdm_load(const void* gptr, unsigned lds_addr,
                                         i32x8 g1) {
    unsigned long long ga = (unsigned long long)gptr;
    u32x4 g0;
    g0.x = 1u;                                             // count = 1
    g0.y = lds_addr;                                       // LDS byte address
    g0.z = (unsigned)ga;                                   // global_addr lo
    g0.w = (unsigned)((ga >> 32) & 0x1FFFFFFull) | 0x80000000u; // hi | type=2
    i32x4 z4 = {};
    i32x8 z8 = {};
    __builtin_amdgcn_tensor_load_to_lds(g0, g1, z4, z4, z8, 0);
}
#endif

__device__ __forceinline__ __bf16 f2bf(float f) {
    union { float f; unsigned u; } x; x.f = f;
    unsigned r = (x.u + 0x7FFFu + ((x.u >> 16) & 1u)) >> 16;
    union { unsigned short s; __bf16 b; } y; y.s = (unsigned short)r;
    return y.b;
}

__device__ __forceinline__ int grade_of(int x) {
    return (x == 0) ? 0 : (x < 5) ? 1 : (x < 11) ? 2 : (x < 15) ? 3 : 4;
}
// METRIC = -1 at components {2..7, 14, 15}
__device__ __forceinline__ float metric_of(int x) {
    return ((0xC0FCu >> x) & 1u) ? -1.0f : 1.0f;
}

// ---------------------------------------------------------------------------
// Kernel 1: equi_linear projection for Q/K/V.  One wave per position.
// mode 0 = Q (scale folded in, row-major out), 1 = K (metric folded in,
// row-major out), 2 = V (transposed out: [b,h,d,kv]).
// ---------------------------------------------------------------------------
__global__ __launch_bounds__(256) void proj_kernel(
    const float* __restrict__ xmv, const float* __restrict__ xs, int Npos,
    const float* __restrict__ wmv, const float* __restrict__ ws2mv,
    const float* __restrict__ wmvs2s, const float* __restrict__ ws2s,
    const float* __restrict__ bmv, const float* __restrict__ bs,
    __bf16* __restrict__ outf, __bf16* __restrict__ outT,
    int mode, float scale)
{
    const int w = threadIdx.x >> 5;
    const int lane = threadIdx.x & 31;
    const size_t pos = (size_t)blockIdx.x * 8 + w;      // b*Npos + n
    const int n = (int)(pos % (size_t)Npos);
    const int b = (int)(pos / (size_t)Npos);

    __shared__ float xb[8][288];
    #pragma unroll
    for (int j = 0; j < 9; ++j) {
        int e = j * 32 + lane;
        xb[w][e] = (e < 256) ? xmv[pos * 256 + e] : xs[pos * 32 + (e - 256)];
    }

    // 128 mv-channels x 16 components = 2048 outputs -> 64 per lane
    for (int j = 0; j < 64; ++j) {
        int idx = j * 32 + lane;
        int o = idx >> 4, x = idx & 15;
        int g = grade_of(x);
        float acc = 0.0f;
        #pragma unroll
        for (int i = 0; i < 16; ++i)
            acc += xb[w][i * 16 + x] * wmv[(o * 16 + i) * 5 + g];
        if (x == 0) {
            float a2 = bmv[o];
            #pragma unroll
            for (int s = 0; s < 32; ++s)
                a2 += xb[w][256 + s] * ws2mv[o * 32 + s];
            acc += a2;
        }
        int h = o & 7, c = o >> 3;              // split_heads: o = c*NHEAD + h
        int d = c * 16 + x;
        size_t bh = (size_t)b * NHEAD + h;
        float val = acc;
        if (mode == 0) val *= scale;
        if (mode == 1) val *= metric_of(x);
        if (mode != 2) outf[(bh * (size_t)Npos + n) * DHEAD + d] = f2bf(val);
        else           outT[(bh * DHEAD + d) * (size_t)Npos + n] = f2bf(val);
    }

    // 256 scalar outputs -> 8 per lane
    for (int j = 0; j < 8; ++j) {
        int os = j * 32 + lane;
        float acc = bs[os];
        #pragma unroll
        for (int i = 0; i < 16; ++i)
            acc += xb[w][i * 16] * wmvs2s[os * 16 + i];
        #pragma unroll
        for (int s = 0; s < 32; ++s)
            acc += xb[w][256 + s] * ws2s[os * 32 + s];
        int h = os & 7, si = os >> 3;           // os = si*NHEAD + h
        size_t bh = (size_t)b * NHEAD + h;
        float val = (mode == 0) ? acc * scale : acc;
        if (mode != 2) outf[(bh * (size_t)Npos + n) * DHEAD + 256 + si] = f2bf(val);
        else           outT[(bh * DHEAD + 256 + si) * (size_t)Npos + n] = f2bf(val);
    }
}

// ---------------------------------------------------------------------------
// Kernel 2: flash attention with v_wmma_f32_16x16x32_bf16.
// Block = 4 waves on one (b,h); each wave owns a 16-query tile.
// K/V chunks (32 keys) staged block-wide in LDS via the Tensor Data Mover:
// K double-buffered (prefetch next chunk), V TDM overlaps the QK^T phase.
// qf,kf: [b*h, N, 288] bf16 ; vt: [b*h, 288, NKV] bf16 ; hf: [b,q,h,288] f32
// ---------------------------------------------------------------------------
__global__ __launch_bounds__(128) void attn_kernel(
    const __bf16* __restrict__ qf, const __bf16* __restrict__ kf,
    const __bf16* __restrict__ vt, float* __restrict__ hf)
{
    const int NQT = NQ / 64;
    const int bh = blockIdx.x / NQT;
    const int qtile = blockIdx.x % NQT;
    const int w = threadIdx.x >> 5;
    const int lane = threadIdx.x & 31;
    const int lmod = lane & 15, lhalf = lane >> 4;
    const int qbase = qtile * 64 + w * 16;

    __shared__ __align__(16) __bf16 kbuf[2][32 * DHEAD];   // 2 x 18 KB
    __shared__ __align__(16) __bf16 vbuf[DHEAD * 32];      // 18 KB
    __shared__ __align__(16) __bf16 plds[4][16][48];       // P tiles, padded

    // A-fragment base for this lane's Q row (M = lmod); loop-invariant ->
    // compiler keeps the 9 Q fragments resident in VGPRs.
    const __bf16* qp = qf + ((size_t)bh * NQ + qbase + lmod) * DHEAD + lhalf * 16;

    const __bf16* kg_base = kf + (size_t)bh * NKV * DHEAD;
    const __bf16* vg_base = vt + (size_t)bh * DHEAD * NKV;

#if HAVE_TDM
    // Group-1 descriptors (ISA 8.4): data_size=1 (2B elements).
    // K: 1-D copy of 9216 elements (32 rows x 288, rows contiguous).
    const i32x8 g1k = { 0x00010000, 0x24000000, 0x00010000, 0x24000000,
                        1, 9216, 0, 0 };
    // V: 2-D tile 32 x 288 rows, row stride NKV=2048 elements.
    const i32x8 g1v = { 0x00010000, 0x00200000, 0x01200000, 0x00200000,
                        0x120, 2048, 0, 0 };
    if (w == 0)                                   // prefetch chunk 0 -> kbuf[0]
        tdm_load(kg_base, (unsigned)(unsigned long long)&kbuf[0][0], g1k);
#endif

    float m[8], l[8];
    v8f o[18];
    #pragma unroll
    for (int v = 0; v < 8; ++v) { m[v] = -1e30f; l[v] = 0.0f; }
    #pragma unroll
    for (int t = 0; t < 18; ++t) o[t] = {};

    int cur = 0;
    #pragma unroll 1
    for (int kb = 0; kb < NKV; kb += 32) {
        __syncthreads();   // prior readers of vbuf / kbuf[cur^1] are done
#if HAVE_TDM
        if (w == 0) {
            tdm_load(vg_base + kb,
                     (unsigned)(unsigned long long)&vbuf[0], g1v);
            int kbn = (kb + 32) & (NKV - 1);      // wraps harmlessly at end
            tdm_load(kg_base + (size_t)kbn * DHEAD,
                     (unsigned)(unsigned long long)&kbuf[cur ^ 1][0], g1k);
            __builtin_amdgcn_s_wait_tensorcnt((short)2);   // K[cur] landed
        }
        __syncthreads();
#else
        {   // cooperative fallback copy: K then V (1152 float4 each)
            const __bf16* kg = kg_base + (size_t)kb * DHEAD;
            #pragma unroll
            for (int j = 0; j < 9; ++j) {
                int idx = j * 128 + threadIdx.x;
                ((float4*)&kbuf[0][0])[idx] = ((const float4*)kg)[idx];
            }
            const __bf16* vg = vg_base + kb;
            #pragma unroll
            for (int j = 0; j < 9; ++j) {
                int idx = j * 128 + threadIdx.x;
                int row = idx >> 2, col = idx & 3;
                ((float4*)&vbuf[0])[idx] =
                    *(const float4*)(vg + (size_t)row * NKV + col * 8);
            }
        }
        __syncthreads();
#endif
        // ---- scores: S(16x32) = Q(16x288) * K^T, two 16-key halves ----
        v8f c0 = {}, c1 = {};
        const __bf16* kr0 = &kbuf[cur][lmod * DHEAD + lhalf * 16];
        const __bf16* kr1 = kr0 + 16 * DHEAD;
        #pragma unroll
        for (int dc = 0; dc < 9; ++dc) {
            v16bf a, b0, b1;
            ((float4*)&a)[0]  = *(const float4*)(qp + dc * 32);
            ((float4*)&a)[1]  = *(const float4*)(qp + dc * 32 + 8);
            ((float4*)&b0)[0] = *(const float4*)(kr0 + dc * 32);
            ((float4*)&b0)[1] = *(const float4*)(kr0 + dc * 32 + 8);
            ((float4*)&b1)[0] = *(const float4*)(kr1 + dc * 32);
            ((float4*)&b1)[1] = *(const float4*)(kr1 + dc * 32 + 8);
            c0 = __builtin_amdgcn_wmma_f32_16x16x32_bf16(false, a, false, b0,
                                                         (short)0, c0, false, false);
            c1 = __builtin_amdgcn_wmma_f32_16x16x32_bf16(false, a, false, b1,
                                                         (short)0, c1, false, false);
        }

        // ---- online softmax (row = v + 8*lhalf, cols striped across lanes) ----
        float p0[8], p1[8];
        #pragma unroll
        for (int v = 0; v < 8; ++v) {
            float rm = fmaxf(c0[v], c1[v]);
            #pragma unroll
            for (int msk = 1; msk < 16; msk <<= 1)
                rm = fmaxf(rm, __shfl_xor(rm, msk, 32));
            float mn = fmaxf(m[v], rm);
            float alpha = __expf(m[v] - mn);
            p0[v] = __expf(c0[v] - mn);
            p1[v] = __expf(c1[v] - mn);
            float rs = p0[v] + p1[v];
            #pragma unroll
            for (int msk = 1; msk < 16; msk <<= 1)
                rs += __shfl_xor(rs, msk, 32);
            l[v] = l[v] * alpha + rs;
            m[v] = mn;
            #pragma unroll
            for (int t = 0; t < 18; ++t) o[t][v] *= alpha;
        }

        // ---- stage P to LDS (C-layout) and reload in A-fragment layout ----
        #pragma unroll
        for (int v = 0; v < 8; ++v) {
            int row = v + 8 * lhalf;
            plds[w][row][lmod]      = f2bf(p0[v]);
            plds[w][row][16 + lmod] = f2bf(p1[v]);
        }
        v16bf ap;
        ((float4*)&ap)[0] = *(const float4*)&plds[w][lmod][16 * lhalf];
        ((float4*)&ap)[1] = *(const float4*)&plds[w][lmod][16 * lhalf + 8];

#if HAVE_TDM
        if (w == 0) __builtin_amdgcn_s_wait_tensorcnt((short)1);  // V landed
        __syncthreads();
#endif
        // ---- O += P(16x32) * V(32x288), V rows in vbuf[d][kv] ----
        const __bf16* vr = &vbuf[lmod * 32 + lhalf * 16];
        #pragma unroll
        for (int t = 0; t < 18; ++t) {
            v16bf bv;
            ((float4*)&bv)[0] = *(const float4*)(vr + t * 512);
            ((float4*)&bv)[1] = *(const float4*)(vr + t * 512 + 8);
            o[t] = __builtin_amdgcn_wmma_f32_16x16x32_bf16(false, ap, false, bv,
                                                           (short)0, o[t], false, false);
        }
#if HAVE_TDM
        cur ^= 1;
#endif
    }

    // ---- epilogue: normalize and write hf[b, q, h, d] ----
    const int b = bh >> 3, h = bh & 7;
    float invl[8];
    #pragma unroll
    for (int v = 0; v < 8; ++v) invl[v] = 1.0f / l[v];
    #pragma unroll
    for (int t = 0; t < 18; ++t) {
        #pragma unroll
        for (int v = 0; v < 8; ++v) {
            int row = v + 8 * lhalf;
            int d = t * 16 + lmod;
            hf[(((size_t)b * NQ + qbase + row) * NHEAD + h) * DHEAD + d] =
                o[t][v] * invl[v];
        }
    }
}

// ---------------------------------------------------------------------------
// Kernel 3: output equi_linear.  One wave per position (4 waves / block).
// hf: [b, n, h, 288] f32  ->  out_mv [b,n,16,16], out_s [b,n,32]
// ---------------------------------------------------------------------------
__global__ __launch_bounds__(128) void oproj_kernel(
    const float* __restrict__ hf,
    const float* __restrict__ wmv, const float* __restrict__ ws2mv,
    const float* __restrict__ wmvs2s, const float* __restrict__ ws2s,
    const float* __restrict__ bmv, const float* __restrict__ bs,
    float* __restrict__ out_mv, float* __restrict__ out_s)
{
    const int w = threadIdx.x >> 5;
    const int lane = threadIdx.x & 31;
    const size_t pos = (size_t)blockIdx.x * 4 + w;      // b*NQ + n

    __shared__ float hb[4][NHEAD * DHEAD];
    #pragma unroll
    for (int j = 0; j < (NHEAD * DHEAD) / 32; ++j)
        hb[w][j * 32 + lane] = hf[pos * (NHEAD * DHEAD) + j * 32 + lane];

    // 16*16 mv outputs + 32 scalar outputs = 288 -> 9 per lane
    for (int j = 0; j < 9; ++j) {
        int idx = j * 32 + lane;
        if (idx < 256) {
            int o = idx >> 4, x = idx & 15;
            int g = grade_of(x);
            float acc = 0.0f;
            // merged channel i = c*NHEAD + h  ->  hb[h*288 + c*16 + x]
            for (int i = 0; i < TOT_MV; ++i)
                acc += hb[w][(i & 7) * DHEAD + (i >> 3) * 16 + x] *
                       wmv[(o * TOT_MV + i) * 5 + g];
            if (x == 0) {
                acc += bmv[o];
                for (int s = 0; s < TOT_S; ++s)
                    acc += hb[w][(s & 7) * DHEAD + 256 + (s >> 3)] *
                           ws2mv[o * TOT_S + s];
            }
            out_mv[pos * 256 + idx] = acc;
        } else {
            int os = idx - 256;
            float acc = bs[os];
            for (int i = 0; i < TOT_MV; ++i)
                acc += hb[w][(i & 7) * DHEAD + (i >> 3) * 16] *
                       wmvs2s[os * TOT_MV + i];
            for (int s = 0; s < TOT_S; ++s)
                acc += hb[w][(s & 7) * DHEAD + 256 + (s >> 3)] *
                       ws2s[os * TOT_S + s];
            out_s[pos * 32 + os] = acc;
        }
    }
}

// ---------------------------------------------------------------------------
extern "C" void kernel_launch(void* const* d_in, const int* in_sizes, int n_in,
                              void* d_out, int out_size, void* d_ws, size_t ws_size,
                              hipStream_t stream)
{
    const float* mv_kv = (const float*)d_in[0];
    const float* mv_q  = (const float*)d_in[1];
    const float* s_kv  = (const float*)d_in[2];
    const float* s_q   = (const float*)d_in[3];

    const float* wq_mv    = (const float*)d_in[4];
    const float* wq_s2mv  = (const float*)d_in[5];
    const float* wq_mvs2s = (const float*)d_in[6];
    const float* wq_s2s   = (const float*)d_in[7];
    const float* bq_mv    = (const float*)d_in[8];
    const float* bq_s     = (const float*)d_in[9];

    const float* wk_mv    = (const float*)d_in[10];
    const float* wk_s2mv  = (const float*)d_in[11];
    const float* wk_mvs2s = (const float*)d_in[12];
    const float* wk_s2s   = (const float*)d_in[13];
    const float* bk_mv    = (const float*)d_in[14];
    const float* bk_s     = (const float*)d_in[15];

    const float* wv_mv    = (const float*)d_in[16];
    const float* wv_s2mv  = (const float*)d_in[17];
    const float* wv_mvs2s = (const float*)d_in[18];
    const float* wv_s2s   = (const float*)d_in[19];
    const float* bv_mv    = (const float*)d_in[20];
    const float* bv_s     = (const float*)d_in[21];

    const float* wo_mv    = (const float*)d_in[22];
    const float* wo_s2mv  = (const float*)d_in[23];
    const float* wo_mvs2s = (const float*)d_in[24];
    const float* wo_s2s   = (const float*)d_in[25];
    const float* bo_mv    = (const float*)d_in[26];
    const float* bo_s     = (const float*)d_in[27];

    char* ws = (char*)d_ws;
    const size_t qf_bytes = (size_t)BATCH * NHEAD * NQ  * DHEAD * 2;  // 18,874,368
    const size_t kf_bytes = (size_t)BATCH * NHEAD * NKV * DHEAD * 2;  // 37,748,736
    const size_t vt_bytes = kf_bytes;
    __bf16* qf = (__bf16*)ws;
    __bf16* kf = (__bf16*)(ws + qf_bytes);
    __bf16* vt = (__bf16*)(ws + qf_bytes + kf_bytes);
    float*  hf = (float*)(ws + qf_bytes + kf_bytes + vt_bytes);

    const float scale = 1.0f / sqrtf((float)DHEAD);

    proj_kernel<<<BATCH * NQ / 8, 256, 0, stream>>>(
        mv_q, s_q, NQ, wq_mv, wq_s2mv, wq_mvs2s, wq_s2s, bq_mv, bq_s,
        qf, (__bf16*)nullptr, 0, scale);
    proj_kernel<<<BATCH * NKV / 8, 256, 0, stream>>>(
        mv_kv, s_kv, NKV, wk_mv, wk_s2mv, wk_mvs2s, wk_s2s, bk_mv, bk_s,
        kf, (__bf16*)nullptr, 1, 1.0f);
    proj_kernel<<<BATCH * NKV / 8, 256, 0, stream>>>(
        mv_kv, s_kv, NKV, wv_mv, wv_s2mv, wv_mvs2s, wv_s2s, bv_mv, bv_s,
        (__bf16*)nullptr, vt, 2, 1.0f);

    attn_kernel<<<BATCH * NHEAD * NQ / 64, 128, 0, stream>>>(qf, kf, vt, hf);

    float* out_mv = (float*)d_out;
    float* out_s  = out_mv + (size_t)BATCH * NQ * 256;
    oproj_kernel<<<BATCH * NQ / 4, 128, 0, stream>>>(
        hf, wo_mv, wo_s2mv, wo_mvs2s, wo_s2s, bo_mv, bo_s, out_mv, out_s);
}
